// SuperGlueOnnxWrapper_670014898630
// MI455X (gfx1250) — compile-verified, matrix operand order checked by reading, not run on hardware
//
#include <hip/hip_runtime.h>
#include <hip/hip_bf16.h>
#include <math.h>

// ---------------------------------------------------------------------------
// SuperGlue forward for MI455X (gfx1250, wave32, WMMA).
// All dense GEMMs use v_wmma_f32_16x16x32_f16 (f16 in, f32 accumulate).
// Working set (<80MB) is L2-resident (192MB) -> no LDS staging in GEMMs.
// ---------------------------------------------------------------------------

typedef _Float16 f16;
typedef _Float16 v8h  __attribute__((ext_vector_type(8)));
typedef _Float16 v16h __attribute__((ext_vector_type(16)));
typedef float    v8f  __attribute__((ext_vector_type(8)));

#define NPTS   1024
#define DD     256
#define NH     4
#define DHD    64
#define NL     18
#define OT_IT  20
#define THRV   0.2f
#define EPSV   1e-5f
#define ZLD    1025   // coupling-matrix row stride (1025x1025)

// ---------------------------------------------------------------------------
// WMMA helpers
// ---------------------------------------------------------------------------
__device__ inline v8f wmma32(v16h a, v16h b, v8f c) {
  // v_wmma_f32_16x16x32_f16  D = A*B + C
  return __builtin_amdgcn_wmma_f32_16x16x32_f16(false, a, false, b, (short)0, c,
                                                false, false);
}

// Load one 16x32 (f16) fragment from a per-lane base pointer.
// Per CDNA5 ISA layout (16-bit A 16x32): lanes 0-15 hold row (row0+lane),
// K = {kb..kb+7, kb+16..kb+23}; lanes 16-31 hold K = {kb+8..15, kb+24..31}.
// The lane-dependent +kh*8 part is baked into the base pointer.
__device__ inline v16h frag_ld(const f16* __restrict__ p) {
  union { v16h v; v8h h[2]; } u;
  u.h[0] = *(const v8h*)p;          // K = kb + kh*8 .. +7
  u.h[1] = *(const v8h*)(p + 16);   // K = kb + 16 + kh*8 .. +7
  return u.v;
}

// ---------------------------------------------------------------------------
// Generic fused-epilogue GEMM:  Y[M x N] = scale * (A[M x K] * B[N x K]^T)
// MODE 0: std epilogue (bias / BN+relu / residual / f32 / f16 stores)
// MODE 1: q,k head split  -> out16[(o&3)][m][o>>2]      ([H][NPTS][DH])
// MODE 2: v head split^T  -> out16[(o&3)][o>>2][m]      ([H][DH][NPTS])
// ---------------------------------------------------------------------------
struct Epi {
  const float* bias;                       // null => none
  const float* g; const float* be;         // BN (g!=null => apply)
  const float* mu; const float* var;
  int   relu;
  float scale;
  float* resid; int ldr;                   // acc += resid[m*ldr+n]; store back
  float* out32; int ldo32;                 // optional f32 store
  f16*   out16; int ldo16; int cs16; int co16; // out16[m*ldo16 + n*cs16 + co16]
};

template<int MODE>
__device__ inline void epi_tile(const v8f& c, int mb, int n, const Epi& e) {
#pragma unroll
  for (int r = 0; r < 8; ++r) {
    const int m = mb + r;
    float acc = c[r] * e.scale;
    if (MODE == 0) {
      if (e.bias) acc += e.bias[n];
      if (e.g)    acc = (acc - e.mu[n]) * (e.g[n] * rsqrtf(e.var[n] + EPSV)) + e.be[n];
      if (e.relu) acc = fmaxf(acc, 0.0f);
      if (e.resid) { acc += e.resid[(size_t)m * e.ldr + n];
                     e.resid[(size_t)m * e.ldr + n] = acc; }
      if (e.out32) e.out32[(size_t)m * e.ldo32 + n] = acc;
      if (e.out16) e.out16[(size_t)m * e.ldo16 + n * e.cs16 + e.co16] = (f16)acc;
    } else if (MODE == 1) {        // q/k: channel o -> (h=o&3, dh=o>>2)
      acc += e.bias[n];
      e.out16[(size_t)(n & 3) * (NPTS * DHD) + (size_t)m * DHD + (n >> 2)] = (f16)acc;
    } else {                       // v transposed: [h][dh][m]
      acc += e.bias[n];
      e.out16[(size_t)(n & 3) * (DHD * NPTS) + (size_t)(n >> 2) * NPTS + m] = (f16)acc;
    }
  }
}

template<int MODE>
__global__ __launch_bounds__(256) void k_gemm(const f16* __restrict__ A, int lda,
                                              const f16* __restrict__ B, int ldb,
                                              int M, int N, int K, Epi e) {
  const int lane = threadIdx.x & 31;
  const int wave = threadIdx.x >> 5;
  const int tilesN = N >> 5;
  const int tile = blockIdx.x * 8 + wave;
  if (tile >= (M >> 5) * tilesN) return;      // wave-uniform (EXEC stays all-1s)
  const int m0 = (tile / tilesN) << 5;
  const int n0 = (tile % tilesN) << 5;

  // Hoisted per-lane fragment pointers; each K-chunk advances by 32 f16 (64B),
  // which the compiler folds into global_load immediate offsets after unroll.
  const int lr = lane & 15;
  const int kh = ((lane >> 4) & 1) * 8;
  const f16* pA0 = A + (size_t)(m0 + lr) * lda + kh;
  const f16* pA1 = pA0 + (size_t)16 * lda;
  const f16* pB0 = B + (size_t)(n0 + lr) * ldb + kh;
  const f16* pB1 = pB0 + (size_t)16 * ldb;

  v8f c00 = {}, c01 = {}, c10 = {}, c11 = {};
  for (int k = 0; k < K; k += 32) {
    // Speculative L2 prefetch of the next chunk (OOB prefetch is dropped).
    __builtin_prefetch(pA0 + 32, 0, 1);
    __builtin_prefetch(pB0 + 32, 0, 1);
    v16h a0 = frag_ld(pA0);
    v16h a1 = frag_ld(pA1);
    v16h b0 = frag_ld(pB0);
    v16h b1 = frag_ld(pB1);
    c00 = wmma32(a0, b0, c00);
    c01 = wmma32(a0, b1, c01);
    c10 = wmma32(a1, b0, c10);
    c11 = wmma32(a1, b1, c11);
    pA0 += 32; pA1 += 32; pB0 += 32; pB1 += 32;
  }
  // C/D layout: lanes 0-15 -> M=r, N=lane; lanes 16-31 -> M=8+r, N=lane-16
  const int lm = ((lane >> 4) & 1) << 3;
  const int ln = lane & 15;
  epi_tile<MODE>(c00, m0 + lm,      n0 + ln,      e);
  epi_tile<MODE>(c01, m0 + lm,      n0 + 16 + ln, e);
  epi_tile<MODE>(c10, m0 + 16 + lm, n0 + ln,      e);
  epi_tile<MODE>(c11, m0 + 16 + lm, n0 + 16 + ln, e);
}

// ---------------------------------------------------------------------------
// Small utility kernels
// ---------------------------------------------------------------------------
__global__ void k_cvt16(const float* __restrict__ s, f16* __restrict__ d, int n) {
  int i = blockIdx.x * blockDim.x + threadIdx.x;
  if (i < n) d[i] = (f16)s[i];
}

__global__ void k_build_x(const float* __restrict__ kpts,
                          const float* __restrict__ sc, float* __restrict__ X) {
  int n = blockIdx.x * blockDim.x + threadIdx.x;
  if (n < NPTS) {
    X[n * 3 + 0] = kpts[n * 2 + 0];
    X[n * 3 + 1] = kpts[n * 2 + 1];
    X[n * 3 + 2] = sc[n];
  }
}

// point-wise conv1d (small channel counts), optional BN+relu
__global__ void k_pconv(const float* __restrict__ X, const float* __restrict__ W,
                        const float* __restrict__ b, const float* g,
                        const float* be, const float* mu, const float* var,
                        float* __restrict__ Y, int ci, int co) {
  int idx = blockIdx.x * blockDim.x + threadIdx.x;
  if (idx >= NPTS * co) return;
  int n = idx / co, o = idx % co;
  const float* x = X + (size_t)n * ci;
  const float* w = W + (size_t)o * ci;
  float acc = b[o];
  for (int i = 0; i < ci; ++i) acc += x[i] * w[i];
  if (g) {
    acc = (acc - mu[o]) * (g[o] * rsqrtf(var[o] + EPSV)) + be[o];
    acc = fmaxf(acc, 0.0f);
  }
  Y[(size_t)n * co + o] = acc;
}

// final kenc layer fused with desc-transpose-add; writes f32 residual stream
// and the f16 GEMM-input copy (first half of the 512-wide concat buffer).
__global__ void k_kenc_final(const float* __restrict__ X, const float* __restrict__ W,
                             const float* __restrict__ b, const float* __restrict__ desc,
                             float* __restrict__ d32, f16* __restrict__ cat16) {
  int idx = blockIdx.x * blockDim.x + threadIdx.x;
  if (idx >= NPTS * DD) return;
  int n = idx / DD, o = idx % DD;
  const float* x = X + (size_t)n * 128;
  const float* w = W + (size_t)o * 128;
  float acc = b[o];
  for (int i = 0; i < 128; ++i) acc += x[i] * w[i];
  acc += desc[(size_t)o * NPTS + n];     // desc is [D][N]
  d32[(size_t)n * DD + o] = acc;
  cat16[(size_t)n * 512 + o] = (f16)acc;
}

// row softmax: S (f32, [rows x n]) -> P (f16)
__global__ __launch_bounds__(256) void k_softmax(const float* __restrict__ S,
                                                 f16* __restrict__ P, int n) {
  const int row = blockIdx.x;
  const float* s = S + (size_t)row * n;
  __shared__ float red[256];
  float mx = -1e30f;
  for (int j = threadIdx.x; j < n; j += 256) mx = fmaxf(mx, s[j]);
  red[threadIdx.x] = mx; __syncthreads();
  for (int o = 128; o; o >>= 1) {
    if (threadIdx.x < o) red[threadIdx.x] = fmaxf(red[threadIdx.x], red[threadIdx.x + o]);
    __syncthreads();
  }
  mx = red[0]; __syncthreads();
  float sum = 0.f;
  for (int j = threadIdx.x; j < n; j += 256) sum += __expf(s[j] - mx);
  red[threadIdx.x] = sum; __syncthreads();
  for (int o = 128; o; o >>= 1) {
    if (threadIdx.x < o) red[threadIdx.x] += red[threadIdx.x + o];
    __syncthreads();
  }
  const float inv = 1.0f / red[0];
  for (int j = threadIdx.x; j < n; j += 256)
    P[(size_t)row * n + j] = (f16)(__expf(s[j] - mx) * inv);
}

// Sinkhorn setup: dustbin row/col/corner = alpha, u = v = 0
__global__ void k_init_bins(float* __restrict__ Z, const float* __restrict__ alphap,
                            float* __restrict__ u, float* __restrict__ v) {
  int i = blockIdx.x * blockDim.x + threadIdx.x;
  float a = *alphap;
  if (i < NPTS) { Z[(size_t)i * ZLD + NPTS] = a; Z[(size_t)NPTS * ZLD + i] = a; }
  if (i == NPTS) Z[(size_t)NPTS * ZLD + NPTS] = a;
  if (i < ZLD) { u[i] = 0.f; v[i] = 0.f; }
}

// u[i] = log_mu[i] - LSE_j( Z[i][j] + v[j] )
__global__ __launch_bounds__(256) void k_ot_u(const float* __restrict__ Z,
                                              const float* __restrict__ v,
                                              float* __restrict__ u) {
  const int i = blockIdx.x;
  const float* z = Z + (size_t)i * ZLD;
  __shared__ float red[256];
  float mx = -1e30f;
  for (int j = threadIdx.x; j < ZLD; j += 256) mx = fmaxf(mx, z[j] + v[j]);
  red[threadIdx.x] = mx; __syncthreads();
  for (int o = 128; o; o >>= 1) {
    if (threadIdx.x < o) red[threadIdx.x] = fmaxf(red[threadIdx.x], red[threadIdx.x + o]);
    __syncthreads();
  }
  mx = red[0]; __syncthreads();
  float s = 0.f;
  for (int j = threadIdx.x; j < ZLD; j += 256) s += __expf(z[j] + v[j] - mx);
  red[threadIdx.x] = s; __syncthreads();
  for (int o = 128; o; o >>= 1) {
    if (threadIdx.x < o) red[threadIdx.x] += red[threadIdx.x + o];
    __syncthreads();
  }
  if (threadIdx.x == 0) {
    const float norm = -logf(2048.0f);
    const float lmu  = (i < NPTS) ? norm : (logf(1024.0f) + norm);
    u[i] = lmu - (mx + logf(red[0]));
  }
}

// v[j] = log_nu[j] - LSE_i( Z[i][j] + u[i] )
__global__ __launch_bounds__(256) void k_ot_v(const float* __restrict__ Z,
                                              const float* __restrict__ u,
                                              float* __restrict__ v) {
  const int j = blockIdx.x;
  __shared__ float red[256];
  float mx = -1e30f;
  for (int i = threadIdx.x; i < ZLD; i += 256)
    mx = fmaxf(mx, Z[(size_t)i * ZLD + j] + u[i]);
  red[threadIdx.x] = mx; __syncthreads();
  for (int o = 128; o; o >>= 1) {
    if (threadIdx.x < o) red[threadIdx.x] = fmaxf(red[threadIdx.x], red[threadIdx.x + o]);
    __syncthreads();
  }
  mx = red[0]; __syncthreads();
  float s = 0.f;
  for (int i = threadIdx.x; i < ZLD; i += 256)
    s += __expf(Z[(size_t)i * ZLD + j] + u[i] - mx);
  red[threadIdx.x] = s; __syncthreads();
  for (int o = 128; o; o >>= 1) {
    if (threadIdx.x < o) red[threadIdx.x] += red[threadIdx.x + o];
    __syncthreads();
  }
  if (threadIdx.x == 0) {
    const float norm = -logf(2048.0f);
    const float lnu  = (j < NPTS) ? norm : (logf(1024.0f) + norm);
    v[j] = lnu - (mx + logf(red[0]));
  }
}

// argmax/max over interior of final transport matrix (tie -> lowest index)
template<int COL>
__global__ __launch_bounds__(256) void k_axmax(const float* __restrict__ Z,
                                               const float* __restrict__ u,
                                               const float* __restrict__ v,
                                               float* __restrict__ mx_out,
                                               int* __restrict__ ix_out) {
  const int a = blockIdx.x;                       // fixed row (or col)
  const float norm = -logf(2048.0f);
  float best = -1e30f; int bj = 0;
  for (int b = threadIdx.x; b < NPTS; b += 256) {
    float val = COL ? (Z[(size_t)b * ZLD + a] + u[b] + v[a] - norm)
                    : (Z[(size_t)a * ZLD + b] + u[a] + v[b] - norm);
    if (val > best) { best = val; bj = b; }
  }
  __shared__ float bm[256]; __shared__ int bi[256];
  bm[threadIdx.x] = best; bi[threadIdx.x] = bj; __syncthreads();
  for (int o = 128; o; o >>= 1) {
    if (threadIdx.x < o) {
      float mo = bm[threadIdx.x + o]; int io = bi[threadIdx.x + o];
      if (mo > bm[threadIdx.x] || (mo == bm[threadIdx.x] && io < bi[threadIdx.x])) {
        bm[threadIdx.x] = mo; bi[threadIdx.x] = io;
      }
    }
    __syncthreads();
  }
  if (threadIdx.x == 0) { mx_out[a] = bm[0]; ix_out[a] = bi[0]; }
}

__global__ void k_match0(const float* __restrict__ max0, const int* __restrict__ idx0,
                         const int* __restrict__ idx1, float* __restrict__ m0,
                         float* __restrict__ ms0) {
  int i = blockIdx.x * blockDim.x + threadIdx.x;
  if (i >= NPTS) return;
  int j = idx0[i];
  bool mutual = (idx1[j] == i);
  float ms = mutual ? expf(max0[i]) : 0.0f;
  ms0[i] = ms;
  m0[i] = (mutual && ms > THRV) ? (float)j : -1.0f;
}

__global__ void k_match1(const int* __restrict__ idx0, const int* __restrict__ idx1,
                         const float* __restrict__ ms0, const float* __restrict__ m0,
                         float* __restrict__ m1, float* __restrict__ ms1) {
  int j = blockIdx.x * blockDim.x + threadIdx.x;
  if (j >= NPTS) return;
  int i = idx1[j];
  bool mutual = (idx0[i] == j);
  ms1[j] = mutual ? ms0[i] : 0.0f;
  m1[j] = (mutual && m0[i] >= 0.0f) ? (float)i : -1.0f;
}

// ---------------------------------------------------------------------------
// Host orchestration
// ---------------------------------------------------------------------------
#define GEMM_GRID(M, N) (((M) >> 5) * ((N) >> 5) / 8)

extern "C" void kernel_launch(void* const* d_in, const int* in_sizes, int n_in,
                              void* d_out, int out_size, void* d_ws, size_t ws_size,
                              hipStream_t stream) {
  (void)in_sizes; (void)n_in; (void)out_size; (void)ws_size;

  // ---- inputs (setup_inputs() order, nested containers flattened in order) --
  const float* kpts0 = (const float*)d_in[0];
  const float* kpts1 = (const float*)d_in[1];
  const float* desc0 = (const float*)d_in[2];
  const float* desc1 = (const float*)d_in[3];
  const float* sc0   = (const float*)d_in[4];
  const float* sc1   = (const float*)d_in[5];
  const float* kW[4]  = {(const float*)d_in[6],  (const float*)d_in[7],
                         (const float*)d_in[8],  (const float*)d_in[9]};
  const float* kB[4]  = {(const float*)d_in[10], (const float*)d_in[11],
                         (const float*)d_in[12], (const float*)d_in[13]};
  const float* kG[3]  = {(const float*)d_in[14], (const float*)d_in[15],
                         (const float*)d_in[16]};
  const float* kBe[3] = {(const float*)d_in[17], (const float*)d_in[18],
                         (const float*)d_in[19]};
  const float* kMu[3] = {(const float*)d_in[20], (const float*)d_in[21],
                         (const float*)d_in[22]};
  const float* kVa[3] = {(const float*)d_in[23], (const float*)d_in[24],
                         (const float*)d_in[25]};
  const float* Wq = (const float*)d_in[26];  const float* bq = (const float*)d_in[27];
  const float* Wk = (const float*)d_in[28];  const float* bk = (const float*)d_in[29];
  const float* Wv = (const float*)d_in[30];  const float* bv = (const float*)d_in[31];
  const float* Wm = (const float*)d_in[32];  const float* bm = (const float*)d_in[33];
  const float* W0 = (const float*)d_in[34];  const float* b0 = (const float*)d_in[35];
  const float* g0 = (const float*)d_in[36];  const float* be0 = (const float*)d_in[37];
  const float* mu0 = (const float*)d_in[38]; const float* var0 = (const float*)d_in[39];
  const float* W1 = (const float*)d_in[40];  const float* b1 = (const float*)d_in[41];
  const float* Wp = (const float*)d_in[42];  const float* bp = (const float*)d_in[43];
  const float* alphap = (const float*)d_in[44];

  // ---- workspace bump allocator (total ~48 MB) ----------------------------
  char* cur = (char*)d_ws;
  auto alloc = [&](size_t bytes) -> void* {
    void* r = (void*)cur;
    cur += (bytes + 255) & ~(size_t)255;
    return r;
  };

  const int NWQ = NL * DD * DD;          // 1179648 per projection stack
  const int NW0 = NL * 2 * DD * 2 * DD;  // 4718592
  const int NW1 = NL * DD * 2 * DD;      // 2359296

  f16* wq16 = (f16*)alloc((size_t)NWQ * 2);
  f16* wk16 = (f16*)alloc((size_t)NWQ * 2);
  f16* wv16 = (f16*)alloc((size_t)NWQ * 2);
  f16* wm16 = (f16*)alloc((size_t)NWQ * 2);
  f16* w016 = (f16*)alloc((size_t)NW0 * 2);
  f16* w116 = (f16*)alloc((size_t)NW1 * 2);
  f16* wp16 = (f16*)alloc((size_t)DD * DD * 2);

  f16*   cat0 = (f16*)alloc((size_t)NPTS * 512 * 2);   // [n][x(256) | msg(256)]
  f16*   cat1 = (f16*)alloc((size_t)NPTS * 512 * 2);
  float* d032 = (float*)alloc((size_t)NPTS * DD * 4);  // f32 residual stream
  float* d132 = (float*)alloc((size_t)NPTS * DD * 4);
  f16*   q16  = (f16*)alloc((size_t)NH * NPTS * DHD * 2);
  f16*   k16  = (f16*)alloc((size_t)NH * NPTS * DHD * 2);
  f16*   vT16 = (f16*)alloc((size_t)NH * DHD * NPTS * 2);
  float* S32  = (float*)alloc((size_t)NPTS * NPTS * 4);
  f16*   P16  = (f16*)alloc((size_t)NPTS * NPTS * 2);
  f16*   o16  = (f16*)alloc((size_t)NPTS * DD * 2);
  f16*   h16  = (f16*)alloc((size_t)NPTS * 512 * 2);
  f16*   md0  = (f16*)alloc((size_t)NPTS * DD * 2);
  f16*   md1  = (f16*)alloc((size_t)NPTS * DD * 2);
  float* Z    = (float*)alloc((size_t)ZLD * ZLD * 4);
  float* u    = (float*)alloc((size_t)ZLD * 4);
  float* v    = (float*)alloc((size_t)ZLD * 4);
  float* max0 = (float*)alloc((size_t)NPTS * 4);
  int*   idx0 = (int*)alloc((size_t)NPTS * 4);
  float* max1 = (float*)alloc((size_t)NPTS * 4);
  int*   idx1 = (int*)alloc((size_t)NPTS * 4);
  float* x3   = (float*)alloc((size_t)NPTS * 3 * 4);
  float* xA   = (float*)alloc((size_t)NPTS * 128 * 4);
  float* xB   = (float*)alloc((size_t)NPTS * 128 * 4);

  // ---- weight f32 -> f16 conversion ---------------------------------------
  auto cvt = [&](const float* s, f16* d, int n) {
    k_cvt16<<<(n + 255) / 256, 256, 0, stream>>>(s, d, n);
  };
  cvt(Wq, wq16, NWQ); cvt(Wk, wk16, NWQ); cvt(Wv, wv16, NWQ); cvt(Wm, wm16, NWQ);
  cvt(W0, w016, NW0); cvt(W1, w116, NW1); cvt(Wp, wp16, DD * DD);

  // ---- keypoint encoder (scalar; tiny channels) ---------------------------
  const float* kps[2]  = {kpts0, kpts1};
  const float* scs[2]  = {sc0, sc1};
  const float* dsc[2]  = {desc0, desc1};
  float* ds[2]         = {d032, d132};
  f16*   cats[2]       = {cat0, cat1};
  const int kci[3] = {3, 32, 64}, kco[3] = {32, 64, 128};
  for (int img = 0; img < 2; ++img) {
    k_build_x<<<4, 256, 0, stream>>>(kps[img], scs[img], x3);
    const float* xin = x3;
    float* ping[2] = {xA, xB};
    for (int lyr = 0; lyr < 3; ++lyr) {
      k_pconv<<<(NPTS * kco[lyr] + 255) / 256, 256, 0, stream>>>(
          xin, kW[lyr], kB[lyr], kG[lyr], kBe[lyr], kMu[lyr], kVa[lyr],
          ping[lyr & 1], kci[lyr], kco[lyr]);
      xin = ping[lyr & 1];
    }
    k_kenc_final<<<(NPTS * DD + 255) / 256, 256, 0, stream>>>(
        xin, kW[3], kB[3], dsc[img], ds[img], cats[img]);
  }

  // ---- GNN: 18 alternating self/cross attention layers --------------------
  for (int l = 0; l < NL; ++l) {
    const f16* wql = wq16 + (size_t)l * DD * DD;
    const f16* wkl = wk16 + (size_t)l * DD * DD;
    const f16* wvl = wv16 + (size_t)l * DD * DD;
    const f16* wml = wm16 + (size_t)l * DD * DD;
    const f16* w0l = w016 + (size_t)l * 2 * DD * 2 * DD;
    const f16* w1l = w116 + (size_t)l * DD * 2 * DD;
    const float* bql = bq + l * DD, *bkl = bk + l * DD, *bvl = bv + l * DD;
    const float* bml = bm + l * DD;
    const float* b0l = b0 + l * 2 * DD, *g0l = g0 + l * 2 * DD;
    const float* be0l = be0 + l * 2 * DD, *mu0l = mu0 + l * 2 * DD;
    const float* var0l = var0 + l * 2 * DD, *b1l = b1 + l * DD;

    for (int img = 0; img < 2; ++img) {
      const int src = (l % 2 == 0) ? img : (1 - img);
      {  // q = x * Wq^T + bq -> head-split [H][N][DH]
        Epi e{}; e.scale = 1.f; e.bias = bql; e.out16 = q16;
        k_gemm<1><<<GEMM_GRID(NPTS, DD), 256, 0, stream>>>(
            cats[img], 512, wql, DD, NPTS, DD, DD, e);
      }
      {  // k from source descriptors
        Epi e{}; e.scale = 1.f; e.bias = bkl; e.out16 = k16;
        k_gemm<1><<<GEMM_GRID(NPTS, DD), 256, 0, stream>>>(
            cats[src], 512, wkl, DD, NPTS, DD, DD, e);
      }
      {  // v from source, stored transposed per head [H][DH][M]
        Epi e{}; e.scale = 1.f; e.bias = bvl; e.out16 = vT16;
        k_gemm<2><<<GEMM_GRID(NPTS, DD), 256, 0, stream>>>(
            cats[src], 512, wvl, DD, NPTS, DD, DD, e);
      }
      for (int h = 0; h < NH; ++h) {
        {  // S = q_h * k_h^T / sqrt(DH)
          Epi e{}; e.scale = 0.125f; e.out32 = S32; e.ldo32 = NPTS;
          k_gemm<0><<<GEMM_GRID(NPTS, NPTS), 256, 0, stream>>>(
              q16 + (size_t)h * NPTS * DHD, DHD,
              k16 + (size_t)h * NPTS * DHD, DHD, NPTS, NPTS, DHD, e);
        }
        k_softmax<<<NPTS, 256, 0, stream>>>(S32, P16, NPTS);
        {  // O_h = P * v_h  -> interleave back: o[n][4*dh + h]
          Epi e{}; e.scale = 1.f; e.out16 = o16; e.ldo16 = DD;
          e.cs16 = NH; e.co16 = h;
          k_gemm<0><<<GEMM_GRID(NPTS, DHD), 256, 0, stream>>>(
              P16, NPTS, vT16 + (size_t)h * DHD * NPTS, NPTS,
              NPTS, DHD, NPTS, e);
        }
      }
      {  // message = o * Wm^T + bm -> second half of concat buffer (f16)
        Epi e{}; e.scale = 1.f; e.bias = bml;
        e.out16 = cats[img] + DD; e.ldo16 = 512; e.cs16 = 1;
        k_gemm<0><<<GEMM_GRID(NPTS, DD), 256, 0, stream>>>(
            o16, DD, wml, DD, NPTS, DD, DD, e);
      }
    }
    // MLP + residual (after both messages computed: uses pre-update x)
    for (int img = 0; img < 2; ++img) {
      {  // h = relu(BN(cat * W0^T + b0))
        Epi e{}; e.scale = 1.f; e.bias = b0l;
        e.g = g0l; e.be = be0l; e.mu = mu0l; e.var = var0l; e.relu = 1;
        e.out16 = h16; e.ldo16 = 512; e.cs16 = 1;
        k_gemm<0><<<GEMM_GRID(NPTS, 512), 256, 0, stream>>>(
            cats[img], 512, w0l, 512, NPTS, 512, 512, e);
      }
      {  // d += h * W1^T + b1 ; refresh f16 x-copy in concat buffer
        Epi e{}; e.scale = 1.f; e.bias = b1l;
        e.resid = ds[img]; e.ldr = DD;
        e.out16 = cats[img]; e.ldo16 = 512; e.cs16 = 1;
        k_gemm<0><<<GEMM_GRID(NPTS, DD), 256, 0, stream>>>(
            h16, 512, w1l, 512, NPTS, DD, 512, e);
      }
    }
  }

  // ---- final projection + score matrix ------------------------------------
  {
    Epi e{}; e.scale = 1.f; e.bias = bp; e.out16 = md0; e.ldo16 = DD; e.cs16 = 1;
    k_gemm<0><<<GEMM_GRID(NPTS, DD), 256, 0, stream>>>(cat0, 512, wp16, DD,
                                                       NPTS, DD, DD, e);
  }
  {
    Epi e{}; e.scale = 1.f; e.bias = bp; e.out16 = md1; e.ldo16 = DD; e.cs16 = 1;
    k_gemm<0><<<GEMM_GRID(NPTS, DD), 256, 0, stream>>>(cat1, 512, wp16, DD,
                                                       NPTS, DD, DD, e);
  }
  k_init_bins<<<(ZLD + 255) / 256, 256, 0, stream>>>(Z, alphap, u, v);
  {  // scores = mdesc0 . mdesc1^T / sqrt(D), written straight into couplings
    Epi e{}; e.scale = 0.0625f; e.out32 = Z; e.ldo32 = ZLD;
    k_gemm<0><<<GEMM_GRID(NPTS, NPTS), 256, 0, stream>>>(md0, DD, md1, DD,
                                                         NPTS, NPTS, DD, e);
  }

  // ---- Sinkhorn (log-domain) ----------------------------------------------
  for (int it = 0; it < OT_IT; ++it) {
    k_ot_u<<<ZLD, 256, 0, stream>>>(Z, v, u);
    k_ot_v<<<ZLD, 256, 0, stream>>>(Z, u, v);
  }

  // ---- matching extraction ------------------------------------------------
  float* out  = (float*)d_out;
  float* m0o  = out;             // matches0
  float* m1o  = out + NPTS;      // matches1
  float* ms0o = out + 2 * NPTS;  // mscores0
  float* ms1o = out + 3 * NPTS;  // mscores1
  k_axmax<0><<<NPTS, 256, 0, stream>>>(Z, u, v, max0, idx0);
  k_axmax<1><<<NPTS, 256, 0, stream>>>(Z, u, v, max1, idx1);
  k_match0<<<4, 256, 0, stream>>>(max0, idx0, idx1, m0o, ms0o);
  k_match1<<<4, 256, 0, stream>>>(idx0, idx1, ms0o, m0o, m1o, ms1o);
}
